// EdgePredictor_74801150427199
// MI455X (gfx1250) — matrix operand order repaired and checked by reading, not compiled, not run on hardware
//
#include <hip/hip_runtime.h>
#include <hip/hip_bf16.h>

typedef __attribute__((ext_vector_type(16))) _Float16 v16h;
typedef __attribute__((ext_vector_type(8)))  float    v8f;

#define IN_DIM     128
#define HIDDEN     128
#define KCHUNKS    8            // 256 / 32
#define NTILES     8            // 128 / 16
#define WAVES_PER_BLOCK 8
#define BLOCK_THREADS   (WAVES_PER_BLOCK * 32)

// LDS image of W1 in f16, pre-swizzled into WMMA B-fragment order:
// frag[kc][nt][lane][j] = W1[kc*32 + (j/8)*16 + (lane/16)*8 + (j&7)][nt*16 + (lane&15)]
#define W1_FRAG_ELEMS (KCHUNKS * NTILES * 32 * 16)   // 32768 halfs = 64 KB

__global__ __launch_bounds__(BLOCK_THREADS)
void edge_mlp_wmma_kernel(const float* __restrict__ x,
                          const long long* __restrict__ edge_index,
                          const float* __restrict__ W1,
                          const float* __restrict__ b1,
                          const float* __restrict__ W2,
                          const float* __restrict__ b2,
                          float* __restrict__ out,
                          int n_edges)
{
    __shared__ _Float16 sW1[W1_FRAG_ELEMS];

    const int tid  = threadIdx.x;
    const int lane = tid & 31;
    const int wave = tid >> 5;

    // ---- Stage W1: iterate linearly over W1 (coalesced b32 reads),
    //      scatter f16 into the fragment-ordered LDS image. ----
    for (int idx = tid; idx < 2 * IN_DIM * HIDDEN; idx += BLOCK_THREADS) {
        const int k  = idx >> 7;            // W1 row   (K = 0..255)
        const int n  = idx & 127;           // W1 col   (N = 0..127)
        const int kc = k >> 5;
        const int j  = (k & 7) | (((k >> 4) & 1) << 3);
        const int l  = (((k >> 3) & 1) << 4) | (n & 15);
        const int nt = n >> 4;
        sW1[((kc * NTILES + nt) * 32 + l) * 16 + j] = (_Float16)W1[idx];
    }
    __syncthreads();

    const int tile    = blockIdx.x * WAVES_PER_BLOCK + wave;
    const int n_tiles = (n_edges + 15) >> 4;
    if (tile >= n_tiles) return;

    // ---- Per-lane edge gather setup (lane L and L+16: same edge, disjoint K) ----
    const int e_local = lane & 15;
    int e = tile * 16 + e_local;
    if (e >= n_edges) e = n_edges - 1;                  // clamp ragged tail (reads only)
    const int rowid = (int)edge_index[e];
    const int colid = (int)edge_index[n_edges + e];
    const float* __restrict__ xr = x + (long)rowid * IN_DIM;
    const float* __restrict__ xc = x + (long)colid * IN_DIM;
    const int khigh = (lane >> 4) << 3;                 // 0 or 8

    v8f acc[NTILES];
#pragma unroll
    for (int nt = 0; nt < NTILES; ++nt) acc[nt] = (v8f){0.f,0.f,0.f,0.f,0.f,0.f,0.f,0.f};

    // ---- Layer 1, fully software-pipelined:
    //      * A gathers: prefetch chunk kc+1 under chunk kc's WMMAs
    //      * B frags:   double-buffered; sched_barrier(0) pins ds_load(t+1)
    //                   ahead of wmma(t) so the LDS latency hides under it ----
    float4 p0, p1, p2, p3;                              // A prefetch buffer (16 floats)
    {
        const int off = khigh;                          // kc = 0 -> x[row]
        p0 = *(const float4*)(xr + off);
        p1 = *(const float4*)(xr + off + 4);
        p2 = *(const float4*)(xr + off + 16);
        p3 = *(const float4*)(xr + off + 20);
    }

    v16h bcur = *(const v16h*)&sW1[(0 * 32 + lane) * 16];   // frag (kc=0, nt=0)

#pragma unroll
    for (int kc = 0; kc < KCHUNKS; ++kc) {
        // Convert the prefetched chunk into the A fragment (16-bit A layout).
        v16h a;
        a[0]  = (_Float16)p0.x;  a[1]  = (_Float16)p0.y;
        a[2]  = (_Float16)p0.z;  a[3]  = (_Float16)p0.w;
        a[4]  = (_Float16)p1.x;  a[5]  = (_Float16)p1.y;
        a[6]  = (_Float16)p1.z;  a[7]  = (_Float16)p1.w;
        a[8]  = (_Float16)p2.x;  a[9]  = (_Float16)p2.y;
        a[10] = (_Float16)p2.z;  a[11] = (_Float16)p2.w;
        a[12] = (_Float16)p3.x;  a[13] = (_Float16)p3.y;
        a[14] = (_Float16)p3.z;  a[15] = (_Float16)p3.w;

        // Issue next chunk's gather now; it completes under the WMMAs below.
        if (kc < KCHUNKS - 1) {
            const int kn = kc + 1;
            const float* __restrict__ p = (kn < 4) ? xr : xc;
            const int off = (kn & 3) * 32 + khigh;      // 32B-aligned
            p0 = *(const float4*)(p + off);
            p1 = *(const float4*)(p + off + 4);
            p2 = *(const float4*)(p + off + 16);
            p3 = *(const float4*)(p + off + 20);
        }

#pragma unroll
        for (int nt = 0; nt < NTILES; ++nt) {
            // Prefetch the NEXT B fragment, then fence the scheduler so the
            // ds_load cannot be sunk below the WMMA that consumes the
            // PREVIOUS fragment.
            const int t = kc * NTILES + nt + 1;
            v16h bnext;
            if (t < KCHUNKS * NTILES) {
                bnext = *(const v16h*)&sW1[(t * 32 + lane) * 16];
            } else {
                bnext = bcur;
            }
            __builtin_amdgcn_sched_barrier(0);
            acc[nt] = __builtin_amdgcn_wmma_f32_16x16x32_f16(
                false, a, false, bcur, (short)0, acc[nt], false, false);
            __builtin_amdgcn_sched_barrier(0);
            bcur = bnext;
        }
    }

    // ---- Epilogue: +b1, ReLU, dot with W2, +b2, sigmoid ----
    const int n_local = lane & 15;
    float b1v[NTILES], w2v[NTILES];
#pragma unroll
    for (int nt = 0; nt < NTILES; ++nt) {
        b1v[nt] = b1[nt * 16 + n_local];
        w2v[nt] = W2[nt * 16 + n_local];
    }
    const float bias2 = b2[0];

#pragma unroll
    for (int r = 0; r < 8; ++r) {
        // C layout: acc[nt][r] = h[M][N], M = r + 8*(lane/16), N = nt*16 + (lane&15)
        float s = 0.f;
#pragma unroll
        for (int nt = 0; nt < NTILES; ++nt) {
            float v = acc[nt][r] + b1v[nt];
            v = v > 0.f ? v : 0.f;
            s = fmaf(v, w2v[nt], s);
        }
        // reduce across the 16 lanes of each half-wave (wave32)
        s += __shfl_xor(s, 1, 32);
        s += __shfl_xor(s, 2, 32);
        s += __shfl_xor(s, 4, 32);
        s += __shfl_xor(s, 8, 32);

        if (n_local == 0) {
            const int eo = tile * 16 + r + ((lane >> 4) << 3);
            if (eo < n_edges)
                out[eo] = 1.0f / (1.0f + __expf(-(s + bias2)));
        }
    }
}

extern "C" void kernel_launch(void* const* d_in, const int* in_sizes, int n_in,
                              void* d_out, int out_size, void* d_ws, size_t ws_size,
                              hipStream_t stream) {
    const float*     x  = (const float*)d_in[0];
    const long long* ei = (const long long*)d_in[1];   // int64 [2, E]
    const float*     W1 = (const float*)d_in[2];       // [256, 128]
    const float*     b1 = (const float*)d_in[3];       // [128]
    const float*     W2 = (const float*)d_in[4];       // [128, 1]
    const float*     b2 = (const float*)d_in[5];       // [1]
    float*           out = (float*)d_out;

    const int n_edges = in_sizes[1] / 2;
    const int n_tiles = (n_edges + 15) / 16;
    const int blocks  = (n_tiles + WAVES_PER_BLOCK - 1) / WAVES_PER_BLOCK;

    hipLaunchKernelGGL(edge_mlp_wmma_kernel, dim3(blocks), dim3(BLOCK_THREADS), 0, stream,
                       x, ei, W1, b1, W2, b2, out, n_edges);
}
